// GATLayer_19258633356001
// MI455X (gfx1250) — matrix operand order repaired and checked by reading, not compiled, not run on hardware
//
#include <hip/hip_runtime.h>

#define N_NODES 8192
#define F 128
#define LRELU_ALPHA 0.2f

#define NCHUNK 16                    // j-dimension chunks
#define CHUNK (N_NODES / NCHUNK)     // 512 j per wave
#define ROWTILES (N_NODES / 16)      // 512

typedef __attribute__((ext_vector_type(16))) __bf16 v16bf;
typedef __attribute__((ext_vector_type(8)))  float  v8f;

__device__ __forceinline__ float lrelu(float x) { return x >= 0.f ? x : LRELU_ALPHA * x; }

// float -> bf16, round-to-nearest-even (avoid relying on fptrunc legality)
__device__ __forceinline__ __bf16 f2bf(float f) {
    unsigned u = __builtin_bit_cast(unsigned, f);
    unsigned r = (u + 0x7FFFu + ((u >> 16) & 1u)) >> 16;
    unsigned short s = (unsigned short)r;
    return __builtin_bit_cast(__bf16, s);
}

// ---------------- zero scratch (acc + lsum contiguous) ----------------
__global__ void gat_zero_kernel(float* __restrict__ p, int n) {
    int i = blockIdx.x * blockDim.x + threadIdx.x;
    if (i < n) p[i] = 0.f;
}

// ---------------- Wh = X @ W ; s1 = Wh.a1 ; s2 = Wh.a2 ; pack Wh to WMMA-B layout ----------------
// B packing used here (contiguous k, same order as A): VGPR v, half h, lane l  <->  K = (l/16)*16 + 2v + h , N = l%16
__global__ void gat_wh_kernel(const float* __restrict__ X, const float* __restrict__ W,
                              const float* __restrict__ a1, const float* __restrict__ a2,
                              __bf16* __restrict__ WhB, float* __restrict__ s1, float* __restrict__ s2) {
    __shared__ float xs[F];
    __shared__ float red[F];
    const int i = blockIdx.x;       // row (also K index of B)
    const int t = threadIdx.x;      // output column
    xs[t] = X[(size_t)i * F + t];
    __syncthreads();
    float acc = 0.f;
#pragma unroll 8
    for (int k = 0; k < F; ++k) acc = fmaf(xs[k], W[k * F + t], acc);

    // pack into B-operand layout: tile (jb = i/32) x colblock (cb = t/16)
    const int cb = t >> 4, nc = t & 15;
    const int jb = i >> 5, kin = i & 31;
    const int lane = ((kin >> 4) << 4) | nc;
    WhB[((size_t)(jb * 8 + cb) * 32 + lane) * 16 + (kin & 15)] = f2bf(acc);

    red[t] = acc * a1[t];
    __syncthreads();
    for (int s = 64; s > 0; s >>= 1) { if (t < s) red[t] += red[t + s]; __syncthreads(); }
    if (t == 0) s1[i] = red[0];
    __syncthreads();
    red[t] = acc * a2[t];
    __syncthreads();
    for (int s = 64; s > 0; s >>= 1) { if (t < s) red[t] += red[t + s]; __syncthreads(); }
    if (t == 0) s2[i] = red[0];
}

// ---------------- global max of s2 ----------------
__global__ void gat_s2max_kernel(const float* __restrict__ s2, float* __restrict__ out) {
    __shared__ float red[256];
    const int t = threadIdx.x;
    float m = -3.0e38f;
    for (int i = t; i < N_NODES; i += 256) m = fmaxf(m, s2[i]);
    red[t] = m;
    __syncthreads();
    for (int s = 128; s > 0; s >>= 1) { if (t < s) red[t] = fmaxf(red[t], red[t + s]); __syncthreads(); }
    if (t == 0) out[0] = red[0];
}

// ---------------- main fused kernel: stream adj once, p = masked exp, acc += p @ Wh (WMMA) ----------------
__global__ void __launch_bounds__(128) gat_attn_kernel(
        const int* __restrict__ adj, const __bf16* __restrict__ WhB,
        const float* __restrict__ s1, const float* __restrict__ s2,
        const float* __restrict__ s2maxp, float* __restrict__ accg, float* __restrict__ lsum) {
    const int wave  = threadIdx.x >> 5;
    const int lane  = threadIdx.x & 31;
    const int row   = lane & 15;            // A row (M) handled by this lane
    const int lhalf = lane >> 4;            // 0 or 1
    const int rowbase = blockIdx.x * 16;
    const int chunk   = blockIdx.y * 4 + wave;
    const int jstart  = chunk * CHUNK;

    const float s1v = s1[rowbase + row];
    const float Mb  = lrelu(s1v + s2maxp[0]);   // safe per-row upper bound of masked logits
    const int* arow = adj + (size_t)(rowbase + row) * N_NODES;
    const v16bf* Bv = (const v16bf*)WhB;

    v8f acc[8];
#pragma unroll
    for (int cb = 0; cb < 8; ++cb) acc[cb] = (v8f){0.f, 0.f, 0.f, 0.f, 0.f, 0.f, 0.f, 0.f};
    float lpart = 0.f;

    const int koff = lhalf * 16;  // this lane covers K = koff..koff+15

#pragma unroll 1
    for (int step = 0; step < CHUNK / 32; ++step) {
        const int j0 = jstart + step * 32;
        const int* ap = arow + j0 + koff;
        __builtin_prefetch(ap + 64, 0, 1);  // speculative prefetch of next adj line
        int4 A0 = *(const int4*)(ap);
        int4 A1 = *(const int4*)(ap + 4);
        int4 A2 = *(const int4*)(ap + 8);
        int4 A3 = *(const int4*)(ap + 12);
        const float* sp = s2 + j0 + koff;
        float4 S0 = *(const float4*)(sp);
        float4 S1 = *(const float4*)(sp + 4);
        float4 S2 = *(const float4*)(sp + 8);
        float4 S3 = *(const float4*)(sp + 12);

        const int   av[16] = {A0.x, A0.y, A0.z, A0.w, A1.x, A1.y, A1.z, A1.w,
                              A2.x, A2.y, A2.z, A2.w, A3.x, A3.y, A3.z, A3.w};
        const float sv[16] = {S0.x, S0.y, S0.z, S0.w, S1.x, S1.y, S1.z, S1.w,
                              S2.x, S2.y, S2.z, S2.w, S3.x, S3.y, S3.z, S3.w};
        v16bf a;
#pragma unroll
        for (int e = 0; e < 16; ++e) {
            float ev = s1v + sv[e];
            ev = ev >= 0.f ? ev : LRELU_ALPHA * ev;     // leaky_relu
            float p = __expf(ev - Mb);                  // <= 1, stable
            p = av[e] > 0 ? p : 0.f;                    // adjacency mask -> exact 0
            lpart += p;
            a[e] = f2bf(p);
        }

        const int jb = j0 >> 5;
#pragma unroll
        for (int cb = 0; cb < 8; ++cb) {
            v16bf b = Bv[(size_t)(jb * 8 + cb) * 32 + lane];
            acc[cb] = __builtin_amdgcn_wmma_f32_16x16x32_bf16(
                false, a, false, b, (short)0, acc[cb], false, false);
        }
    }

    // row-sum of p: combine the two lane halves covering the same row
    lpart += __shfl_xor(lpart, 16);
    if (lhalf == 0) atomicAdd(&lsum[rowbase + row], lpart);

    // C layout: VGPR r -> M = r + 8*lhalf, N = lane%16
#pragma unroll
    for (int cb = 0; cb < 8; ++cb)
#pragma unroll
        for (int r = 0; r < 8; ++r)
            atomicAdd(&accg[(size_t)(rowbase + r + 8 * lhalf) * F + cb * 16 + (lane & 15)], acc[cb][r]);
}

// ---------------- out = elu(acc / l) ----------------
__global__ void gat_final_kernel(const float* __restrict__ accg, const float* __restrict__ lsum,
                                 float* __restrict__ out) {
    int idx = blockIdx.x * blockDim.x + threadIdx.x;
    if (idx < N_NODES * F) {
        float h = accg[idx] / lsum[idx >> 7];
        out[idx] = h > 0.f ? h : (__expf(h) - 1.f);
    }
}

extern "C" void kernel_launch(void* const* d_in, const int* in_sizes, int n_in,
                              void* d_out, int out_size, void* d_ws, size_t ws_size,
                              hipStream_t stream) {
    const float* X   = (const float*)d_in[0];   // mole_out (8192,128) f32
    const int*   adj = (const int*)d_in[1];     // (8192,8192) i32
    const float* W   = (const float*)d_in[2];   // (128,128) f32
    const float* a1  = (const float*)d_in[3];   // (128,) f32
    const float* a2  = (const float*)d_in[4];   // (128,) f32
    float* out = (float*)d_out;

    char* ws = (char*)d_ws;
    __bf16* WhB   = (__bf16*)(ws);                                   // 2 MB packed B operand
    float*  s1    = (float*)(ws + 2097152);                          // 32 KB
    float*  s2    = (float*)(ws + 2097152 + 32768);                  // 32 KB
    float*  s2max = (float*)(ws + 2097152 + 65536);                  // 256 B slot
    float*  accg  = (float*)(ws + 2097152 + 65536 + 256);            // 4 MB
    float*  lsum  = (float*)(ws + 2097152 + 65536 + 256 + 4194304);  // 32 KB (contiguous after accg)

    const int nzero = N_NODES * F + N_NODES;  // accg + lsum
    gat_zero_kernel<<<(nzero + 255) / 256, 256, 0, stream>>>(accg, nzero);
    gat_wh_kernel<<<N_NODES, F, 0, stream>>>(X, W, a1, a2, WhB, s1, s2);
    gat_s2max_kernel<<<1, 256, 0, stream>>>(s2, s2max);
    gat_attn_kernel<<<dim3(ROWTILES, NCHUNK / 4), 128, 0, stream>>>(adj, WhB, s1, s2, s2max, accg, lsum);
    gat_final_kernel<<<(N_NODES * F + 255) / 256, 256, 0, stream>>>(accg, lsum, out);
}
